// TransformerBlock_73220602462358
// MI455X (gfx1250) — matrix operand order repaired and checked by reading, not compiled, not run on hardware
//
#include <hip/hip_runtime.h>
#include <hip/hip_bf16.h>
#include <math.h>

#define DIM    2048
#define NH     16
#define HD     128
#define HIDDEN 8192
#define BATCH  2
#define SEQ    2048
#define MROWS  (BATCH*SEQ)

typedef unsigned short u16;
typedef __attribute__((ext_vector_type(16))) __bf16 v16bf;
typedef __attribute__((ext_vector_type(8)))  __bf16 v8bf;
typedef __attribute__((ext_vector_type(8)))  float  v8f;

__device__ __forceinline__ u16 f2bf(float f) {
  unsigned u = __builtin_bit_cast(unsigned, f);
  unsigned r = ((u >> 16) & 1u) + 0x7fffu;   // round-to-nearest-even
  return (u16)((u + r) >> 16);
}

__device__ __forceinline__ v16bf mk16(v8bf lo, v8bf hi) {
  v16bf r;
#pragma unroll
  for (int i = 0; i < 8; ++i) { r[i] = lo[i]; r[i + 8] = hi[i]; }
  return r;
}

__device__ __forceinline__ v8f wmma_bf16(v16bf a, v16bf b, v8f c) {
  // D = A(16x32 bf16) x B(32x16 bf16) + C(16x16 f32)
  return __builtin_amdgcn_wmma_f32_16x16x32_bf16(false, a, false, b, (short)0, c,
                                                 false, false);
}

// CDNA5 async global->LDS DMA (ASYNCcnt-tracked, no VGPR data path).
// lds_off: per-lane LDS byte address; gsrc: per-lane 64-bit global address.
__device__ __forceinline__ void async_ld_b128(unsigned lds_off, const void* gsrc) {
  asm volatile("global_load_async_to_lds_b128 %0, %1, off"
               :: "v"(lds_off), "v"(gsrc) : "memory");
}
__device__ __forceinline__ void wait_asynccnt0() {
  asm volatile("s_wait_asynccnt 0" ::: "memory");
}

// ---------------------------------------------------------------- elementwise
__global__ void k_f32_to_bf16(const float* __restrict__ in, u16* __restrict__ out, int n) {
  int i = blockIdx.x * 256 + threadIdx.x;
  if (i < n) out[i] = f2bf(in[i]);
}

// ------------------------------------------------------------------- rmsnorm
__global__ __launch_bounds__(256) void k_rmsnorm_bf16(const float* __restrict__ x,
                                                      const float* __restrict__ g,
                                                      u16* __restrict__ out) {
  int row = blockIdx.x;
  const float* xr = x + (size_t)row * DIM;
  float ss = 0.f;
  for (int i = threadIdx.x; i < DIM; i += 256) { float v = xr[i]; ss += v * v; }
#pragma unroll
  for (int off = 16; off; off >>= 1) ss += __shfl_xor(ss, off, 32);
  __shared__ float red[8];
  __shared__ float rn_s;
  if ((threadIdx.x & 31) == 0) red[threadIdx.x >> 5] = ss;
  __syncthreads();
  if (threadIdx.x == 0) {
    float tot = 0.f;
    for (int i = 0; i < 8; ++i) tot += red[i];
    rn_s = rsqrtf(tot / (float)DIM + 1e-6f);
  }
  __syncthreads();
  float rn = rn_s;
  u16* orow = out + (size_t)row * DIM;
  for (int i = threadIdx.x; i < DIM; i += 256) orow[i] = f2bf(xr[i] * rn * g[i]);
}

// --------------------------------------------------------------------- GEMM
// C[M,N] = A[M,K](bf16, K-contig) x W[N,K](bf16, K-contig)^T
// EPI 0: store bf16   EPI 1: store f32 + resid   EPI 2: silu -> bf16
template <int EPI>
__global__ __launch_bounds__(256) void k_gemm_wmma(const u16* __restrict__ A,
                                                   const u16* __restrict__ W,
                                                   const float* __restrict__ resid,
                                                   void* __restrict__ outp,
                                                   int N, int K) {
  __shared__ __align__(16) u16 sA[128 * 32];
  const int lane = threadIdx.x & 31, wave = threadIdx.x >> 5;
  const int ln = lane & 15, hb = lane >> 4;
  const int wm = wave & 3, wn = wave >> 2;      // 4 waves over M, 2 over N
  const int bm = blockIdx.y * 128, bn = blockIdx.x * 128;

  // per-thread staging addresses: thread copies 32B of A tile per k-step
  const int sr = threadIdx.x >> 1, sseg = threadIdx.x & 1;
  const unsigned ldsDst = (unsigned)(size_t)(void*)&sA[sr * 32 + sseg * 16];
  const u16* gA = A + (size_t)(bm + sr) * K + sseg * 16;

  v8f zero = {};
  v8f acc[2][4];
#pragma unroll
  for (int mi = 0; mi < 2; ++mi)
#pragma unroll
    for (int ni = 0; ni < 4; ++ni) acc[mi][ni] = zero;

  for (int k0 = 0; k0 < K; k0 += 32) {
    // async DMA: A[bm..bm+127][k0..k0+31] (8 KB) -> LDS, no VGPR data path
    async_ld_b128(ldsDst,      gA + k0);
    async_ld_b128(ldsDst + 16, gA + k0 + 8);   // next 16B chunk (8 u16)
    if (k0 + 32 < K)
      __builtin_prefetch(gA + k0 + 32, 0, 1);
    wait_asynccnt0();
    __syncthreads();

    // B fragments straight from global: lane = output column, 16 contiguous K elems
    v16bf bF[4];
#pragma unroll
    for (int ni = 0; ni < 4; ++ni) {
      int ncol = bn + wn * 64 + ni * 16 + ln;
      bF[ni] = *reinterpret_cast<const v16bf*>(W + (size_t)ncol * K + k0 + 16 * hb);
    }
    // A fragments from LDS (lane = row; contiguous K, k = 16*hb + s, same as B)
    v16bf aF[2];
#pragma unroll
    for (int mi = 0; mi < 2; ++mi) {
      int ml = wm * 32 + mi * 16 + ln;
      v8bf lo = *reinterpret_cast<const v8bf*>(&sA[ml * 32 + 16 * hb]);
      v8bf hi = *reinterpret_cast<const v8bf*>(&sA[ml * 32 + 16 * hb + 8]);
      aF[mi] = mk16(lo, hi);
    }
#pragma unroll
    for (int mi = 0; mi < 2; ++mi)
#pragma unroll
      for (int ni = 0; ni < 4; ++ni)
        acc[mi][ni] = wmma_bf16(aF[mi], bF[ni], acc[mi][ni]);
    __syncthreads();
  }

#pragma unroll
  for (int mi = 0; mi < 2; ++mi)
#pragma unroll
    for (int ni = 0; ni < 4; ++ni)
#pragma unroll
      for (int v = 0; v < 8; ++v) {
        int row = bm + wm * 32 + mi * 16 + v + 8 * hb;
        int col = bn + wn * 64 + ni * 16 + ln;
        size_t idx = (size_t)row * N + col;
        float val = acc[mi][ni][v];
        if (EPI == 0) {
          ((u16*)outp)[idx] = f2bf(val);
        } else if (EPI == 1) {
          ((float*)outp)[idx] = resid[idx] + val;
        } else {
          float s = val / (1.f + __expf(-val));  // silu
          ((u16*)outp)[idx] = f2bf(s);
        }
      }
}

// -------------------------------------------------- V transpose (per head, d-major)
__global__ void k_transpose_v(const u16* __restrict__ V, u16* __restrict__ Vt) {
  size_t gid = (size_t)blockIdx.x * 256 + threadIdx.x;  // (bh, d, t)
  int t = (int)(gid % SEQ);
  size_t tmp = gid / SEQ;
  int d = (int)(tmp % HD);
  int bh = (int)(tmp / HD);
  int b = bh >> 4, h = bh & 15;
  Vt[gid] = V[((size_t)(b * SEQ + t)) * DIM + h * HD + d];
}

// ------------------------------------------------------------ flash attention
// Q,K: bf16 [B,T,DIM] (head slice contiguous); Vt: bf16 [B*NH][HD][T]; O: bf16 [B,T,DIM]
__global__ __launch_bounds__(128) void k_attn(const u16* __restrict__ Q,
                                              const u16* __restrict__ Km,
                                              const u16* __restrict__ Vt,
                                              u16* __restrict__ O) {
  __shared__ __align__(16) u16 sP[4][16][32];
  const int lane = threadIdx.x & 31, wave = threadIdx.x >> 5;
  const int ln = lane & 15, hb = lane >> 4;
  const int bh = blockIdx.y, b = bh >> 4, h = bh & 15;
  const int q0 = blockIdx.x * 64 + wave * 16;   // 16 query rows per wave
  // exp2-folded scale: softmax uses exp2((s*scale - m)*log2e); fold log2e*scale
  const float scl2 = 0.08838834764831845f * 1.4426950408889634f; // scale*log2(e)
  const float l2e  = 1.4426950408889634f;

  // Q fragments: 16x128 as 4 chunks of 16x32
  v16bf qf[4];
  const u16* qrow = Q + ((size_t)(b * SEQ + q0 + ln)) * DIM + h * HD;
#pragma unroll
  for (int c = 0; c < 4; ++c) {
    v8bf lo = *reinterpret_cast<const v8bf*>(qrow + c * 32 + 16 * hb);
    v8bf hi = *reinterpret_cast<const v8bf*>(qrow + c * 32 + 16 * hb + 8);
    qf[c] = mk16(lo, hi);
  }

  v8f zero = {};
  v8f o[8];
  float m_i[8], l_i[8];   // m,l tracked in log2 domain
#pragma unroll
  for (int j = 0; j < 8; ++j) o[j] = zero;
#pragma unroll
  for (int v = 0; v < 8; ++v) { m_i[v] = -INFINITY; l_i[v] = 0.f; }

  const int kEnd = blockIdx.x * 64 + 64;  // uniform per block (barrier-safe)
  for (int kb = 0; kb < kEnd; kb += 32) {
    // --- S = Q x K^T for 32 keys (two 16x16 tiles) ---
    v8f s0 = zero, s1 = zero;
    const u16* krow0 = Km + ((size_t)(b * SEQ + kb + ln)) * DIM + h * HD;
    const u16* krow1 = Km + ((size_t)(b * SEQ + kb + 16 + ln)) * DIM + h * HD;
#pragma unroll
    for (int c = 0; c < 4; ++c) {
      v16bf kf0 = *reinterpret_cast<const v16bf*>(krow0 + c * 32 + 16 * hb);
      v16bf kf1 = *reinterpret_cast<const v16bf*>(krow1 + c * 32 + 16 * hb);
      s0 = wmma_bf16(qf[c], kf0, s0);
      s1 = wmma_bf16(qf[c], kf1, s1);
    }

    // --- online softmax update in log2 domain (row = v + 8*hb, col = ln) ---
#pragma unroll
    for (int v = 0; v < 8; ++v) {
      int row = q0 + v + 8 * hb;
      int key0 = kb + ln, key1 = kb + 16 + ln;
      float a0 = s0[v] * scl2 + (key0 > row ? -1e9f * l2e : 0.f);
      float a1 = s1[v] * scl2 + (key1 > row ? -1e9f * l2e : 0.f);
      float t = fmaxf(a0, a1);
#pragma unroll
      for (int off = 1; off < 16; off <<= 1) t = fmaxf(t, __shfl_xor(t, off, 16));
      float mnew = fmaxf(m_i[v], t);
      float e0 = exp2f(a0 - mnew);
      float e1 = exp2f(a1 - mnew);
      float rs = e0 + e1;
#pragma unroll
      for (int off = 1; off < 16; off <<= 1) rs += __shfl_xor(rs, off, 16);
      float alpha = exp2f(m_i[v] - mnew);
      l_i[v] = l_i[v] * alpha + rs;
      m_i[v] = mnew;
#pragma unroll
      for (int j = 0; j < 8; ++j) o[j][v] *= alpha;
      sP[wave][v + 8 * hb][ln] = f2bf(e0);
      sP[wave][v + 8 * hb][16 + ln] = f2bf(e1);
    }
    __syncthreads();

    // --- reshape P (C-layout -> A-fragment) via LDS, then O += P x V ---
    v8bf plo = *reinterpret_cast<const v8bf*>(&sP[wave][ln][16 * hb]);
    v8bf phi = *reinterpret_cast<const v8bf*>(&sP[wave][ln][16 * hb + 8]);
    v16bf pA = mk16(plo, phi);
#pragma unroll
    for (int j = 0; j < 8; ++j) {
      int d = j * 16 + ln;
      v16bf vF = *reinterpret_cast<const v16bf*>(Vt + ((size_t)bh * HD + d) * SEQ + kb + 16 * hb);
      o[j] = wmma_bf16(pA, vF, o[j]);
    }
    __syncthreads();
  }

#pragma unroll
  for (int j = 0; j < 8; ++j)
#pragma unroll
    for (int v = 0; v < 8; ++v) {
      int row = q0 + v + 8 * hb;
      float val = o[j][v] / l_i[v];
      O[((size_t)(b * SEQ + row)) * DIM + h * HD + j * 16 + ln] = f2bf(val);
    }
}

// -------------------------------------------------------------------- launch
extern "C" void kernel_launch(void* const* d_in, const int* in_sizes, int n_in,
                              void* d_out, int out_size, void* d_ws, size_t ws_size,
                              hipStream_t stream) {
  (void)in_sizes; (void)n_in; (void)out_size; (void)ws_size;
  const float* x      = (const float*)d_in[0];
  /* d_in[1] = mask, causal mask applied analytically */
  const float* wq     = (const float*)d_in[2];
  const float* wk     = (const float*)d_in[3];
  const float* wv     = (const float*)d_in[4];
  const float* wo     = (const float*)d_in[5];
  const float* w1     = (const float*)d_in[6];
  const float* w2     = (const float*)d_in[7];
  const float* g_attn = (const float*)d_in[8];
  const float* g_ffn  = (const float*)d_in[9];
  float* out = (float*)d_out;

  char* p = (char*)d_ws;
  auto take = [&](size_t bytes) -> char* {
    char* r = p;
    p += (bytes + 255) & ~(size_t)255;
    return r;
  };
  u16* wq_b  = (u16*)take((size_t)DIM * DIM * 2);
  u16* wk_b  = (u16*)take((size_t)DIM * DIM * 2);
  u16* wv_b  = (u16*)take((size_t)DIM * DIM * 2);
  u16* wo_b  = (u16*)take((size_t)DIM * DIM * 2);
  u16* w1_b  = (u16*)take((size_t)HIDDEN * DIM * 2);
  u16* w2_b  = (u16*)take((size_t)DIM * HIDDEN * 2);
  u16* hbf   = (u16*)take((size_t)MROWS * DIM * 2);
  u16* qbf   = (u16*)take((size_t)MROWS * DIM * 2);
  u16* kbf   = (u16*)take((size_t)MROWS * DIM * 2);
  u16* vbf   = (u16*)take((size_t)MROWS * DIM * 2);
  u16* vtb   = (u16*)take((size_t)MROWS * DIM * 2);
  u16* aob   = (u16*)take((size_t)MROWS * DIM * 2);
  float* x2  = (float*)take((size_t)MROWS * DIM * 4);
  u16* h2bf  = (u16*)take((size_t)MROWS * DIM * 2);
  u16* ubf   = (u16*)take((size_t)MROWS * HIDDEN * 2);

  auto cvt = [&](const float* src, u16* dst, size_t n) {
    k_f32_to_bf16<<<(unsigned)((n + 255) / 256), 256, 0, stream>>>(src, dst, (int)n);
  };
  cvt(wq, wq_b, (size_t)DIM * DIM);
  cvt(wk, wk_b, (size_t)DIM * DIM);
  cvt(wv, wv_b, (size_t)DIM * DIM);
  cvt(wo, wo_b, (size_t)DIM * DIM);
  cvt(w1, w1_b, (size_t)HIDDEN * DIM);
  cvt(w2, w2_b, (size_t)DIM * HIDDEN);

  // 1) h = rmsnorm(x) * g_attn  (bf16)
  k_rmsnorm_bf16<<<MROWS, 256, 0, stream>>>(x, g_attn, hbf);

  // 2) Q/K/V projections
  dim3 gD(DIM / 128, MROWS / 128);
  k_gemm_wmma<0><<<gD, 256, 0, stream>>>(hbf, wq_b, nullptr, qbf, DIM, DIM);
  k_gemm_wmma<0><<<gD, 256, 0, stream>>>(hbf, wk_b, nullptr, kbf, DIM, DIM);
  k_gemm_wmma<0><<<gD, 256, 0, stream>>>(hbf, wv_b, nullptr, vbf, DIM, DIM);

  // 3) V -> per-head d-major
  k_transpose_v<<<(unsigned)(((size_t)MROWS * DIM) / 256), 256, 0, stream>>>(vbf, vtb);

  // 4) flash attention
  dim3 gA(SEQ / 64, BATCH * NH);
  k_attn<<<gA, 128, 0, stream>>>(qbf, kbf, vtb, aob);

  // 5) x2 = x + attnO @ Wo^T
  k_gemm_wmma<1><<<gD, 256, 0, stream>>>(aob, wo_b, x, x2, DIM, DIM);

  // 6) h2 = rmsnorm(x2) * g_ffn
  k_rmsnorm_bf16<<<MROWS, 256, 0, stream>>>(x2, g_ffn, h2bf);

  // 7) u = silu(h2 @ W1^T)  (bf16)
  dim3 gH(HIDDEN / 128, MROWS / 128);
  k_gemm_wmma<2><<<gH, 256, 0, stream>>>(h2bf, w1_b, nullptr, ubf, HIDDEN, DIM);

  // 8) out = x2 + u @ W2^T  (fp32)
  k_gemm_wmma<1><<<gD, 256, 0, stream>>>(ubf, w2_b, x2, out, DIM, HIDDEN);
}
